// MultiHeadUnionAttention_27736898798341
// MI455X (gfx1250) — matrix operand order repaired and checked by reading, not compiled, not run on hardware
//
#include <hip/hip_runtime.h>
#include <stdint.h>

#define H_ 8
#define F_ 16
#define C_ 256
#define B_ 4
#define P_ 16384
#define G_ 4096

typedef __attribute__((ext_vector_type(16))) __bf16 v16bf;
typedef __attribute__((ext_vector_type(8)))  float  v8f;
typedef __attribute__((ext_vector_type(2)))  float  v2f;

union FragBF { v16bf v; uint4 q[2]; };

static __device__ inline v8f zero_v8f() {
  v8f z = {0.f,0.f,0.f,0.f,0.f,0.f,0.f,0.f};
  return z;
}

// ---------------------------------------------------------------------------
// Transpose x (B,C,P) f32 -> xT (B,P,C) bf16, LDS-tiled 32x32.
// ---------------------------------------------------------------------------
__global__ void k_transpose_x(const float* __restrict__ x, __bf16* __restrict__ xT) {
  __shared__ float tile[32][33];
  int b  = blockIdx.z;
  int c0 = blockIdx.y * 32;
  int p0 = blockIdx.x * 32;
  int tx = threadIdx.x, ty = threadIdx.y; // block (32,8)
#pragma unroll
  for (int i = 0; i < 4; ++i) {
    int c = c0 + ty + i * 8;
    tile[ty + i * 8][tx] = x[((size_t)b * C_ + c) * P_ + p0 + tx];
  }
  __syncthreads();
#pragma unroll
  for (int i = 0; i < 4; ++i) {
    int p = p0 + ty + i * 8;
    xT[((size_t)b * P_ + p) * C_ + c0 + tx] = (__bf16)tile[tx][ty + i * 8];
  }
}

// ---------------------------------------------------------------------------
// Pack kv (2 branches, 152x256 -> padded 160x256) and uw (256x256) to bf16.
// ---------------------------------------------------------------------------
__global__ void k_prep_weights(const float* __restrict__ kv0, const float* __restrict__ kv1,
                               const float* __restrict__ uw,
                               __bf16* __restrict__ kvp, __bf16* __restrict__ uwb) {
  int idx = blockIdx.x * 256 + threadIdx.x;
  if (idx < 2 * 160 * 256) {
    int br = idx / (160 * 256);
    int r  = (idx / 256) % 160;
    int c  = idx % 256;
    const float* kv = br ? kv1 : kv0;
    kvp[idx] = (__bf16)((r < 152) ? kv[r * 256 + c] : 0.f);
  }
  if (idx < 256 * 256) uwb[idx] = (__bf16)uw[idx];
}

__global__ void k_zero(float* __restrict__ p, int n) {
  int i = blockIdx.x * 256 + threadIdx.x;
  if (i < n) p[i] = 0.f;
}

// ---------------------------------------------------------------------------
// bf16 WMMA GEMM: C[mat] (M x P) = A[mat/aDiv] (M x 256) * B[mat%4]^T
// B stored transposed as (P x 256) bf16 so K is contiguous.
// One wave per block; wave tile = 32(M) x 64(N); K loop 8 x 32.
// Fragment layouts follow CDNA5 ISA 7.12.2 (wave32).
// ---------------------------------------------------------------------------
__global__ void __launch_bounds__(32)
k_wmma_gemm(const __bf16* __restrict__ A, const __bf16* __restrict__ Bm,
            __bf16* __restrict__ Cm, int M, int aDiv) {
  int lane = threadIdx.x;
  int half = lane >> 4;   // 0: lanes 0-15, 1: lanes 16-31
  int l15  = lane & 15;
  int mat  = blockIdx.z;
  int aIdx = mat / aDiv;
  int bIdx = mat % 4;     // batch
  int m0   = blockIdx.y * 32;
  int p0   = blockIdx.x * 64;

  const __bf16* Abase = A  + (size_t)aIdx * M  * C_;
  const __bf16* Bbase = Bm + (size_t)bIdx * P_ * C_;
  __bf16*       Cbase = Cm + (size_t)mat  * M  * P_;

  const uint4* arow0 = (const uint4*)(Abase + (size_t)(m0 + l15)      * C_);
  const uint4* arow1 = (const uint4*)(Abase + (size_t)(m0 + 16 + l15) * C_);

  v8f acc[2][4];
#pragma unroll
  for (int i = 0; i < 2; ++i)
#pragma unroll
    for (int j = 0; j < 4; ++j) acc[i][j] = zero_v8f();

#pragma unroll
  for (int ks = 0; ks < 8; ++ks) {          // K = 256 in steps of 32
    int qb = ks * 4;                        // uint4 index into 256-bf16 row
    FragBF a0, a1;
    // A 16x32 bf16 layout: lanes<16 K pairs {0..7,16..23}, lanes>=16 {8..15,24..31}
    a0.q[0] = arow0[qb + half];     a0.q[1] = arow0[qb + 2 + half];
    a1.q[0] = arow1[qb + half];     a1.q[1] = arow1[qb + 2 + half];
#pragma unroll
    for (int nt = 0; nt < 4; ++nt) {
      const uint4* brow = (const uint4*)(Bbase + (size_t)(p0 + nt * 16 + l15) * C_);
      FragBF bf;
      // B 32x16 bf16 layout: lanes<16 K {0..7,16..23}, lanes>=16 {8..15,24..31}
      bf.q[0] = brow[qb + half];
      bf.q[1] = brow[qb + 2 + half];
      acc[0][nt] = __builtin_amdgcn_wmma_f32_16x16x32_bf16(
          false, a0.v, false, bf.v, (short)0, acc[0][nt], false, false);
      acc[1][nt] = __builtin_amdgcn_wmma_f32_16x16x32_bf16(
          false, a1.v, false, bf.v, (short)0, acc[1][nt], false, false);
    }
  }
  // C/D layout: VGPR r -> M = r + 8*half, N = l15
#pragma unroll
  for (int mi = 0; mi < 2; ++mi)
#pragma unroll
    for (int nt = 0; nt < 4; ++nt)
#pragma unroll
      for (int r = 0; r < 8; ++r) {
        int m = m0 + mi * 16 + half * 8 + r;
        int p = p0 + nt * 16 + l15;
        Cbase[(size_t)m * P_ + p] = (__bf16)acc[mi][nt][r];
      }
}

// ---------------------------------------------------------------------------
// Per-channel mean / rstd over (nMat, P). data layout (nMat, nch, P) bf16.
// One 256-thread block per channel.
// ---------------------------------------------------------------------------
__global__ void k_stats(const __bf16* __restrict__ data, float* __restrict__ stats,
                        int nch, int nMat) {
  __shared__ float ssum[256], ssq[256];
  int c = blockIdx.x;
  int t = threadIdx.x;
  float s = 0.f, q = 0.f;
  for (int m = 0; m < nMat; ++m) {
    const __bf16* row = data + ((size_t)m * nch + c) * P_;
    for (int p = t; p < P_; p += 256) {
      float v = (float)row[p];
      s += v; q += v * v;
    }
  }
  ssum[t] = s; ssq[t] = q;
  __syncthreads();
  for (int off = 128; off > 0; off >>= 1) {
    if (t < off) { ssum[t] += ssum[t + off]; ssq[t] += ssq[t + off]; }
    __syncthreads();
  }
  if (t == 0) {
    float n = (float)nMat * (float)P_;
    float mean = ssum[0] / n;
    float var  = ssq[0] / n - mean * mean;
    stats[c * 2]     = mean;
    stats[c * 2 + 1] = rsqrtf(var + 1e-5f);
  }
}

// ---------------------------------------------------------------------------
// Lattice scatter: BN keys, tanh lattice, trilinear-ish weight, atomic
// segment-sum of BN'd values into grid (BH, F, G). Saves w and flat index.
// ---------------------------------------------------------------------------
__global__ void k_scatter(const __bf16* __restrict__ z, const float* __restrict__ stats,
                          const float* __restrict__ pts,
                          const float* __restrict__ kg, const float* __restrict__ kb,
                          const float* __restrict__ vg, const float* __restrict__ vb,
                          const float* __restrict__ tw, const float* __restrict__ tb,
                          float* __restrict__ grid, float* __restrict__ w_arr,
                          int* __restrict__ flat_arr, int S, int D, int br) {
  int idx = blockIdx.x * 256 + threadIdx.x;     // over B*H*P
  int p  = idx & (P_ - 1);
  int bh = idx >> 14;
  int h  = bh & 7;
  int b  = bh >> 3;
  const __bf16* zb = z + ((size_t)(br * 4 + b) * 160) * P_;

  float k3[3];
#pragma unroll
  for (int d = 0; d < 3; ++d) {
    int c = h * 3 + d;
    float raw = (float)zb[(size_t)c * P_ + p];
    float bnv = (raw - stats[c * 2]) * stats[c * 2 + 1] * kg[c] + kb[c];
    k3[d] = pts[((size_t)b * 3 + d) * P_ + p] + bnv;
  }

  float wv = 1.f;
  int flat = 0;
  for (int dd = 0; dd < D; ++dd) {
    float key = tb[h * D + dd];
#pragma unroll
    for (int kk = 0; kk < 3; ++kk) key += tw[(h * D + dd) * 3 + kk] * k3[kk];
    float lat  = tanhf(key);
    float pos  = (lat + 1.f) * 0.5f * (float)(S - 1);
    float cell = rintf(pos);
    cell = fminf(fmaxf(cell, 0.f), (float)(S - 1));
    wv  *= 1.f - fabsf(pos - cell);
    flat = flat * S + (int)cell;
  }
  w_arr[idx]    = wv;
  flat_arr[idx] = flat;

  float* gbh = grid + (size_t)bh * F_ * G_ + flat;
#pragma unroll
  for (int f = 0; f < F_; ++f) {
    int c = 24 + h * 16 + f;
    float raw = (float)zb[(size_t)c * P_ + p];
    float bnv = (raw - stats[c * 2]) * stats[c * 2 + 1] * vg[h * 16 + f] + vb[h * 16 + f];
    atomicAdd(gbh + (size_t)f * G_, bnv * wv);
  }
}

// ---------------------------------------------------------------------------
// Grouped conv over the grid via f32 WMMA 16x16x4:
// out[fo, cells16] += sum_taps sum_fi W_tap(16fo x 4fi) * grid(4fi x 16cells).
// A 16x4 f32 layout: lanes<16 K=0,1; lanes>=16 K=2,3 (M = lane&15).
// B 4x16 f32 layout: mirrored (N = lane&15).
// One wave per (bh, 16-cell tile). SAME padding handled by predication.
// ---------------------------------------------------------------------------
__global__ void __launch_bounds__(32)
k_conv_wmma(const float* __restrict__ grid, const float* __restrict__ cw,
            const float* __restrict__ cb, float* __restrict__ convout,
            int S, int D) {
  int lane = threadIdx.x;
  int half = lane >> 4;
  int l15  = lane & 15;
  int bh   = blockIdx.y;
  int h    = bh & 7;
  int c0   = blockIdx.x * 16;      // 16 contiguous cells, never crossing a row
  int x0   = c0 % S;
  int rest = c0 / S;
  int y0   = rest % S;
  int z0   = rest / S;
  int T    = (D == 2) ? 9 : 27;

  const float* gb = grid + (size_t)bh * F_ * G_;
  v8f acc = zero_v8f();

  for (int t = 0; t < T; ++t) {
    int dz = (D == 3) ? (t / 9 - 1) : 0;
    int dy = (D == 3) ? ((t / 3) % 3 - 1) : (t / 3 - 1);
    int dx = t % 3 - 1;
    int ny = y0 + dy, nz = z0 + dz;
    bool outer = (ny >= 0) && (ny < S) && (D == 2 || ((nz >= 0) && (nz < S)));
    int nbase  = ((D == 3) ? (nz * S + ny) : ny) * S + x0 + dx;
    int nx     = x0 + l15 + dx;
    bool valid = outer && (nx >= 0) && (nx < S);
#pragma unroll
    for (int kc = 0; kc < 4; ++kc) {          // fi in chunks of 4
      int fi0 = kc * 4 + half * 2;
      float a0 = cw[(((size_t)(h * F_ + l15)) * F_ + fi0) * T + t];
      float a1 = cw[(((size_t)(h * F_ + l15)) * F_ + fi0 + 1) * T + t];
      float b0 = 0.f, b1 = 0.f;
      if (valid) {
        const float* g0 = gb + (size_t)fi0 * G_ + (nbase + l15);
        b0 = g0[0];
        b1 = g0[G_];
      }
      v2f av = {a0, a1};
      v2f bv = {b0, b1};
      acc = __builtin_amdgcn_wmma_f32_16x16x4_f32(
          false, av, false, bv, (short)0, acc, false, false);
    }
  }
#pragma unroll
  for (int r = 0; r < 8; ++r) {
    int fo = half * 8 + r;
    convout[((size_t)bh * F_ + fo) * G_ + c0 + l15] = acc[r] + cb[h * F_ + fo];
  }
}

// ---------------------------------------------------------------------------
// Gather conv result at saved indices, re-weight: sliced (br*4+b, 128, P) bf16.
// ---------------------------------------------------------------------------
__global__ void k_gather(const float* __restrict__ convout, const float* __restrict__ w_arr,
                         const int* __restrict__ flat_arr, __bf16* __restrict__ sliced,
                         int br) {
  int idx = blockIdx.x * 256 + threadIdx.x;   // over B*H*P
  int p  = idx & (P_ - 1);
  int bh = idx >> 14;
  int h  = bh & 7;
  int b  = bh >> 3;
  float wv = w_arr[idx];
  int flat = flat_arr[idx];
  const float* cbase = convout + (size_t)bh * F_ * G_ + flat;
  __bf16* sb = sliced + ((size_t)(br * 4 + b) * 128 + h * 16) * P_ + p;
#pragma unroll
  for (int f = 0; f < F_; ++f)
    sb[(size_t)f * P_] = (__bf16)(cbase[(size_t)f * G_] * wv);
}

// ---------------------------------------------------------------------------
// BN + ReLU on sliced, write concatenated transposed cat (B, P, 256) bf16.
// ---------------------------------------------------------------------------
__global__ void k_bn_relu_cat(const __bf16* __restrict__ sliced,
                              const float* __restrict__ stats0, const float* __restrict__ stats1,
                              const float* __restrict__ ag0, const float* __restrict__ ab0,
                              const float* __restrict__ ag1, const float* __restrict__ ab1,
                              __bf16* __restrict__ catT) {
  int idx = blockIdx.x * 256 + threadIdx.x;   // over 2*B*128*P
  int p  = idx & (P_ - 1);
  int t1 = idx >> 14;
  int c  = t1 & 127;
  int t2 = t1 >> 7;
  int b  = t2 & 3;
  int br = t2 >> 2;
  const float* st = br ? stats1 : stats0;
  const float* ag = br ? ag1 : ag0;
  const float* ab = br ? ab1 : ab0;
  float v  = (float)sliced[idx];
  float bn = (v - st[c * 2]) * st[c * 2 + 1] * ag[c] + ab[c];
  bn = fmaxf(bn, 0.f);
  catT[((size_t)b * P_ + p) * 256 + br * 128 + c] = (__bf16)bn;
}

// ---------------------------------------------------------------------------
// Final: out = x + relu(BN(g)).  gz layout (b, 256, P) bf16.
// ---------------------------------------------------------------------------
__global__ void k_final(const float* __restrict__ x, const __bf16* __restrict__ gz,
                        const float* __restrict__ stats_g,
                        const float* __restrict__ ug, const float* __restrict__ ub,
                        float* __restrict__ out) {
  size_t idx = (size_t)blockIdx.x * 256 + threadIdx.x;  // B*256*P
  int c = (int)((idx >> 14) & 255);
  float g  = (float)gz[idx];
  float bn = (g - stats_g[c * 2]) * stats_g[c * 2 + 1] * ug[c] + ub[c];
  out[idx] = x[idx] + fmaxf(bn, 0.f);
}

// ---------------------------------------------------------------------------
extern "C" void kernel_launch(void* const* d_in, const int* in_sizes, int n_in,
                              void* d_out, int out_size, void* d_ws, size_t ws_size,
                              hipStream_t stream) {
  (void)in_sizes; (void)n_in; (void)out_size; (void)ws_size;
  const float* x   = (const float*)d_in[0];
  const float* pts = (const float*)d_in[1];
  // Branch params: base 2 (b0) and 13 (b1): kv,kg,kb,vg,vb,tw,tb,cw,cb,ag,ab
  const float* kv[2] = {(const float*)d_in[2],  (const float*)d_in[13]};
  const float* kg[2] = {(const float*)d_in[3],  (const float*)d_in[14]};
  const float* kb[2] = {(const float*)d_in[4],  (const float*)d_in[15]};
  const float* vg[2] = {(const float*)d_in[5],  (const float*)d_in[16]};
  const float* vb[2] = {(const float*)d_in[6],  (const float*)d_in[17]};
  const float* tw[2] = {(const float*)d_in[7],  (const float*)d_in[18]};
  const float* tb[2] = {(const float*)d_in[8],  (const float*)d_in[19]};
  const float* cw[2] = {(const float*)d_in[9],  (const float*)d_in[20]};
  const float* cbp[2]= {(const float*)d_in[10], (const float*)d_in[21]};
  const float* ag[2] = {(const float*)d_in[11], (const float*)d_in[22]};
  const float* ab[2] = {(const float*)d_in[12], (const float*)d_in[23]};
  const float* uw = (const float*)d_in[24];
  const float* ug = (const float*)d_in[25];
  const float* ub = (const float*)d_in[26];

  char* ws = (char*)d_ws;
  __bf16* xT     = (__bf16*)(ws + 0);                // 33,554,432 B
  __bf16* zbuf   = (__bf16*)(ws + 33554432);         // 41,943,040 B
  float*  grid   = (float*)(ws + 75497472);          // 16,777,216 B (both branches)
  float*  conv   = (float*)(ws + 92274688);          // 16,777,216 B (both branches)
  float*  warr   = (float*)(ws + 109051904);         //  4,194,304 B
  int*    farr   = (int*)(ws + 113246208);           //  4,194,304 B
  __bf16* kvp    = (__bf16*)(ws + 117440512);        //    163,840 B
  __bf16* uwb    = (__bf16*)(ws + 117604352);        //    131,072 B
  float*  statsz = (float*)(ws + 117735424);         //      2,560 B (2 x 160 x 2)
  float*  statss = (float*)(ws + 117739520);         //      2,048 B (2 x 128 x 2)
  float*  statsg = (float*)(ws + 117743616);         //      2,048 B (256 x 2)
  // Aliases (lifetimes do not overlap on the serialized stream):
  __bf16* catT   = xT;                               // xT dead after GEMM1
  __bf16* sliced = zbuf;                             // z dead after scatter
  __bf16* gz     = (__bf16*)grid;                    // grid+conv dead after gather

  const size_t gridBr = (size_t)B_ * H_ * F_ * G_;   // 2,097,152 floats per branch
  const size_t bhp    = (size_t)B_ * H_ * P_;        //   524,288 points per branch

  // 1. x -> xT bf16
  k_transpose_x<<<dim3(P_ / 32, C_ / 32, B_), dim3(32, 8), 0, stream>>>(x, xT);
  // 2. weights -> bf16 (kv padded to 160 rows)
  k_prep_weights<<<320, 256, 0, stream>>>(kv[0], kv[1], uw, kvp, uwb);
  // 3. zero both scatter grids (must happen every call)
  k_zero<<<(int)(2 * gridBr / 256), 256, 0, stream>>>(grid, (int)(2 * gridBr));
  // 4. GEMM1: z = kv * x  (bf16 WMMA), mats = br*4 + b
  k_wmma_gemm<<<dim3(P_ / 64, 160 / 32, 8), 32, 0, stream>>>(kvp, xT, zbuf, 160, 4);
  // 5. BN stats over z per branch (keys + values channels)
  k_stats<<<160, 256, 0, stream>>>(zbuf, statsz, 160, 4);
  k_stats<<<160, 256, 0, stream>>>(zbuf + (size_t)4 * 160 * P_, statsz + 320, 160, 4);
  // 6. lattice scatter per branch
  k_scatter<<<(int)(bhp / 256), 256, 0, stream>>>(zbuf, statsz, pts, kg[0], kb[0], vg[0], vb[0],
      tw[0], tb[0], grid, warr, farr, 64, 2, 0);
  k_scatter<<<(int)(bhp / 256), 256, 0, stream>>>(zbuf, statsz + 320, pts, kg[1], kb[1], vg[1], vb[1],
      tw[1], tb[1], grid + gridBr, warr + bhp, farr + bhp, 16, 3, 1);
  // 7. grouped conv per branch (f32 WMMA 16x16x4)
  k_conv_wmma<<<dim3(G_ / 16, B_ * H_), 32, 0, stream>>>(grid, cw[0], cbp[0], conv, 64, 2);
  k_conv_wmma<<<dim3(G_ / 16, B_ * H_), 32, 0, stream>>>(grid + gridBr, cw[1], cbp[1],
                                                         conv + gridBr, 16, 3);
  // 8. gather per branch
  k_gather<<<(int)(bhp / 256), 256, 0, stream>>>(conv, warr, farr, sliced, 0);
  k_gather<<<(int)(bhp / 256), 256, 0, stream>>>(conv + gridBr, warr + bhp, farr + bhp, sliced, 1);
  // 9. BN stats over sliced per branch
  k_stats<<<128, 256, 0, stream>>>(sliced, statss, 128, 4);
  k_stats<<<128, 256, 0, stream>>>(sliced + (size_t)4 * 128 * P_, statss + 256, 128, 4);
  // 10. BN + ReLU + concat -> catT (B, P, 256) bf16
  k_bn_relu_cat<<<65536, 256, 0, stream>>>(sliced, statss, statss + 256,
      ag[0], ab[0], ag[1], ab[1], catT);
  // 11. GEMM2: g = uw * cat (bf16 WMMA), mats = b
  k_wmma_gemm<<<dim3(P_ / 64, 256 / 32, 4), 32, 0, stream>>>(uwb, catT, gz, 256, 4);
  // 12. BN stats over g
  k_stats<<<256, 256, 0, stream>>>(gz, statsg, 256, 4);
  // 13. residual + BN + ReLU -> out (f32)
  k_final<<<65536, 256, 0, stream>>>(x, gz, statsg, ug, ub, (float*)d_out);
}
